// GraphRec_32469952757839
// MI455X (gfx1250) — matrix operand order repaired and checked by reading, not compiled, not run on hardware
//
#include <hip/hip_runtime.h>
#include <hip/hip_bf16.h>

// ---------------------------------------------------------------------------
// GraphRec forward on MI455X (gfx1250, wave32).
// All dense layers run on v_wmma_f32_16x16x32_bf16; embedding gathers +
// f32->bf16 conversion are fused into vectorized WMMA A-fragment loads so the
// 460800x128 concat inputs are never materialized. For K<=128 the entire
// weight matrix is staged K-transposed in LDS once per block (one barrier),
// and the K-loop is fully unrolled and barrier-free: per slice it is just
// two global_load_b128 (A) + two ds_load_b128 per tile (B) + NT WMMAs.
// All GEMM row counts are multiples of 128 and all K are multiples of 32.
// ---------------------------------------------------------------------------

typedef __attribute__((ext_vector_type(16))) __bf16 v16bf;
typedef __attribute__((ext_vector_type(8)))  __bf16 v8bf;
typedef __attribute__((ext_vector_type(8)))  float  v8f;
typedef __attribute__((ext_vector_type(4)))  float  v4f;

#define B_  512
#define LI_ 30
#define LU_ 30

enum { AM_DIRECT = 0, AM_GATHER2 = 1, AM_GATHERC = 2, AM_CONCAT2 = 3,
       AM_GATHER1 = 4, AM_RATE = 5 };
enum { ACT_NONE = 0, ACT_LEAKY = 1, ACT_RELU = 2 };

__device__ __forceinline__ float  bf2f(__bf16 x) { return (float)x; }
__device__ __forceinline__ __bf16 f2bf(float  x) { return (__bf16)x; }

__device__ __forceinline__ v8bf cvt8(v8f x) {
    v8bf r;
#pragma unroll
    for (int i = 0; i < 8; ++i) r[i] = (__bf16)x[i];
    return r;
}

// ---------------------------------------------------------------------------
// out(N, NT*16) = act( A(N,KK) @ W(KK, NT*16) + bias )
// One wave = 16 rows x (NT*16) cols; block = 8 waves = 128 rows.
// Requires N % 128 == 0 and KK % 32 == 0 (true for every layer here).
// ---------------------------------------------------------------------------
template<int AMODE, int NT, int KK>
__global__ void __launch_bounds__(256)
gemm_wmma_kernel(const __bf16* __restrict__ a0, const __bf16* __restrict__ a1,
                 const float* __restrict__ e0, const float* __restrict__ e1,
                 const int* __restrict__ i0, int i0s,
                 const int* __restrict__ i1, int i1s, int groupL,
                 const float* __restrict__ W, const float* __restrict__ bias,
                 int N, int act,
                 __bf16* __restrict__ out)
{
    constexpr int NCOLS   = NT * 16;
    constexpr bool WHOLEW = (KK <= 128);           // whole-W staging fits LDS
    constexpr int LDSK    = WHOLEW ? KK : 32;      // staged K extent
    __shared__ __bf16 ldsBT[NCOLS * LDSK];         // W transposed: [col][k]

    const int tid  = threadIdx.x;
    const int lane = tid & 31;
    const int wave = tid >> 5;
    const int row0 = (blockIdx.x * 8 + wave) * 16;

    // CDNA5 WMMA bf16 fragment geometry (wave32)
    const int m    = lane & 15;           // A row within tile
    const int khA  = (lane >> 4) * 8;     // A K sub-base (0 or 8)
    const int nloc = lane & 15;           // B/C column within tile
    const int khB  = (lane >> 4) * 8;     // B K sub-base (0 or 8)
    const int row  = row0 + m;

    // per-row gather state, loaded once
    size_t ia = 0, ib = 0;
    float  mval = 1.0f;
    if constexpr (AMODE == AM_GATHER2) {
        ia = (size_t)i0[(size_t)row * i0s];
        ib = (size_t)i1[(size_t)row * i1s];
    } else if constexpr (AMODE == AM_GATHERC) {
        ia   = (size_t)i1[row / groupL];
        mval = (i0[(size_t)row * i0s] > 0) ? 1.0f : 0.0f;
    } else if constexpr (AMODE == AM_GATHER1) {
        ia = (size_t)i0[(size_t)row * i0s];
    } else if constexpr (AMODE == AM_RATE) {
        ib = (size_t)i1[row];
    }

    // 8 contiguous A elements at kb (kb never straddles a 64-elem region)
    auto loadA8 = [&](int kb) -> v8bf {
        if constexpr (AMODE == AM_DIRECT) {
            return *(const v8bf*)(a0 + (size_t)row * KK + kb);
        } else if constexpr (AMODE == AM_CONCAT2) {
            const __bf16* p = (kb < 64) ? a0 + (size_t)row * 64 + kb
                                        : a1 + (size_t)row * 64 + (kb - 64);
            return *(const v8bf*)p;
        } else if constexpr (AMODE == AM_GATHER2) {
            const float* p = (kb < 64) ? e0 + ia * 64 + kb
                                       : e1 + ib * 64 + (kb - 64);
            return cvt8(*(const v8f*)p);
        } else if constexpr (AMODE == AM_GATHERC) {
            v8f x = *(const v8f*)(e0 + ia * 64 + kb);
            return cvt8(x * mval);
        } else if constexpr (AMODE == AM_GATHER1) {
            return cvt8(*(const v8f*)(e0 + ia * 64 + kb));
        } else { // AM_RATE: [h(64) | z(64) | encoder_emb[review](768)]
            if (kb < 128) {
                const __bf16* p = (kb < 64) ? a0 + (size_t)row * 64 + kb
                                            : a1 + (size_t)row * 64 + (kb - 64);
                return *(const v8bf*)p;
            }
            return cvt8(*(const v8f*)(e1 + ib * 768 + (kb - 128)));
        }
    };

    // stage W[kbase:kbase+LDSK][:] transposed into LDS (coalesced float4 reads)
    auto stageW = [&](int kbase) {
#pragma unroll
        for (int idx = 0; idx < LDSK * NCOLS; idx += 1024) {
            int e = idx + tid * 4;
            if (LDSK * NCOLS >= 1024 || e < LDSK * NCOLS) {
                int kr = e / NCOLS, c = e % NCOLS;
                v4f wv = *(const v4f*)(W + (size_t)(kbase + kr) * NCOLS + c);
#pragma unroll
                for (int q = 0; q < 4; ++q)
                    ldsBT[(c + q) * LDSK + kr] = f2bf(wv[q]);
            }
        }
    };

    v8f acc[NT];
    v8f zero = {0.f, 0.f, 0.f, 0.f, 0.f, 0.f, 0.f, 0.f};
#pragma unroll
    for (int t = 0; t < NT; ++t) acc[t] = zero;

    // one k-slice: A fragment (two vector loads) + NT B fragments + NT WMMAs
    auto kslice = [&](int k0, int klds) {
        v8bf alo = loadA8(k0 + khA);
        v8bf ahi = loadA8(k0 + khA + 16);
        v16bf afrag;
#pragma unroll
        for (int i = 0; i < 8; ++i) { afrag[i] = alo[i]; afrag[8 + i] = ahi[i]; }
#pragma unroll
        for (int t = 0; t < NT; ++t) {
            const int col = t * 16 + nloc;
            v8bf blo = *(const v8bf*)&ldsBT[col * LDSK + klds + khB];
            v8bf bhi = *(const v8bf*)&ldsBT[col * LDSK + klds + khB + 16];
            v16bf bfrag;
#pragma unroll
            for (int i = 0; i < 8; ++i) { bfrag[i] = blo[i]; bfrag[8 + i] = bhi[i]; }
            acc[t] = __builtin_amdgcn_wmma_f32_16x16x32_bf16(
                false, afrag, false, bfrag, (short)0, acc[t], false, false);
        }
    };

    if constexpr (WHOLEW) {
        // stage entire W once, then a fully unrolled barrier-free K loop
        stageW(0);
        __syncthreads();
#pragma unroll
        for (int k0 = 0; k0 < KK; k0 += 32) kslice(k0, k0);
    } else {
        // large K (rating head, N=512): per-slice staging
        for (int k0 = 0; k0 < KK; k0 += 32) {
            stageW(k0);
            __syncthreads();
            kslice(k0, 0);
            __syncthreads();
        }
    }

    // epilogue: bias + activation; C elem j -> row row0 + j + 8*(lane>=16)
    const int mbase = (lane >> 4) * 8;
#pragma unroll
    for (int t = 0; t < NT; ++t) {
        const int col = t * 16 + nloc;
        const float bv = bias[col];
#pragma unroll
        for (int j = 0; j < 8; ++j) {
            const int r = row0 + mbase + j;
            float v = acc[t][j] + bv;
            if (act == ACT_LEAKY)     v = (v > 0.0f) ? v : 0.2f * v;
            else if (act == ACT_RELU) v = fmaxf(v, 0.0f);
            out[(size_t)r * NCOLS + col] = f2bf(v);
        }
    }
}

// s[r] = X(r,:) . w + b   (attention score head / final 16->1 layer)
// K % 8 == 0, rows 16B-aligned
__global__ void dot_score_kernel(const __bf16* __restrict__ X,
                                 const float* __restrict__ w,
                                 const float* __restrict__ b,
                                 int N, int K, float* __restrict__ out)
{
    int r = blockIdx.x * blockDim.x + threadIdx.x;
    if (r >= N) return;
    const v8bf* xp = (const v8bf*)(X + (size_t)r * K);
    float s = b[0];
    for (int c = 0; c < (K >> 3); ++c) {
        v8bf x = xp[c];
#pragma unroll
        for (int i = 0; i < 8; ++i) s += bf2f(x[i]) * w[c * 8 + i];
    }
    out[r] = s;
}

// masked exp-softmax over L neighbors + weighted sum of 64-dim value rows
__global__ void attnorm_wsum_kernel(const float* __restrict__ score,
                                    const int* __restrict__ nidx, int nstride,
                                    const __bf16* __restrict__ V, int L,
                                    __bf16* __restrict__ out)
{
    __shared__ float a[32];
    int g = blockIdx.x, d = threadIdx.x;  // 64 threads = feature dim
    int base = g * L;
    if (d < L) {
        int id = nidx[(size_t)(base + d) * nstride];
        a[d] = (id > 0) ? __expf(score[base + d]) : 0.0f;
    }
    __syncthreads();
    float denom = 1e-10f, accv = 0.0f;
    for (int j = 0; j < L; ++j) denom += a[j];
    for (int j = 0; j < L; ++j)
        accv += a[j] * bf2f(V[(size_t)(base + j) * 64 + d]);
    out[(size_t)g * 64 + d] = f2bf(accv / denom);
}

// ---------------------------------------------------------------------------
// host side
// ---------------------------------------------------------------------------
extern "C" void kernel_launch(void* const* d_in, const int* in_sizes, int n_in,
                              void* d_out, int out_size, void* d_ws, size_t ws_size,
                              hipStream_t stream)
{
    (void)in_sizes; (void)n_in; (void)out_size; (void)ws_size;

    // inputs (setup_inputs order), then params as jax pytree leaves
    // (dict keys sorted alphabetically, tuples flattened in order)
    const int* uids   = (const int*)d_in[0];
    const int* iids   = (const int*)d_in[1];
    const int* uitem  = (const int*)d_in[2];   // (B,LI,2) [id, rate]
    const int* uuser  = (const int*)d_in[3];   // (B,LU)
    const int* uuitem = (const int*)d_in[4];   // (B,LU,LI,2)
    const int* iuser  = (const int*)d_in[5];   // (B,LU,2)
    const int* review = (const int*)d_in[6];
    auto P = [&](int i) { return (const float*)d_in[i]; };
    // 7:aggre_items{W,b} 9:aggre_items_s1 11:aggre_neigbors_s2 13:aggre_users_i
    // 15:combine{W1,b1,W2,b2} 19:encoder_emb 20:g_u{W1,b1,W2,b2} 24:g_v{...}
    // 28:im_w1{W,b} 30:item_emb 31:item_users_att_i{W1,b1,W2,b2} 35:rate_emb
    // 36:rate_pred{Wa,ba,Wb,bb,Wc,bc} 42:um_w1 44:um_w4 46:um_w5 48:user_emb
    // 49:user_items_att 53:user_items_att_s1 57:user_users_att_s2
    const float* item_emb = P(30);
    const float* rate_emb = P(35);
    const float* user_emb = P(48);
    const float* enc_emb  = P(19);

    const int E1 = B_ * LI_;          // 15360 user-item edges
    const int E2 = B_ * LU_ * LI_;    // 460800 social edges
    const int E3 = B_ * LU_;          // 15360 social/item-user edges

    char* ws = (char*)d_ws;
    size_t off = 0;
    auto carve = [&](size_t bytes) {
        void* p = ws + off; off += (bytes + 255) & ~(size_t)255; return p;
    };
    __bf16* bufA  = (__bf16*)carve((size_t)E2 * 64 * 2);
    __bf16* bufB  = (__bf16*)carve((size_t)E2 * 64 * 2);
    __bf16* bufC  = (__bf16*)carve((size_t)E2 * 64 * 2);
    __bf16* bufD  = (__bf16*)carve((size_t)E2 * 64 * 2);
    float*  score = (float*)carve((size_t)E2 * 4);
    __bf16* g_pre = (__bf16*)carve((size_t)E3 * 64 * 2);   // wsum outputs
    __bf16* h_oI  = (__bf16*)carve((size_t)E3 * 64 * 2);
    __bf16* h_iI  = (__bf16*)carve((size_t)B_ * 64 * 2);
    __bf16* h_iS  = (__bf16*)carve((size_t)B_ * 64 * 2);
    __bf16* h_hid = (__bf16*)carve((size_t)B_ * 128 * 2);
    __bf16* h_fin = (__bf16*)carve((size_t)B_ * 64 * 2);
    __bf16* z_fin = (__bf16*)carve((size_t)B_ * 64 * 2);
    __bf16* r1    = (__bf16*)carve((size_t)B_ * 64 * 2);
    __bf16* r2    = (__bf16*)carve((size_t)B_ * 16 * 2);

    const __bf16* nb = nullptr; const float* nf = nullptr; const int* ni = nullptr;
    auto grid = [](int N) { return (N + 127) / 128; };

#define GEMM(AM, NT, KK, a0, a1, e0, e1, i0, i0s, i1, i1s, L, Wp, bp, N, act, outp) \
    gemm_wmma_kernel<AM, NT, KK><<<grid(N), 256, 0, stream>>>(                       \
        a0, a1, e0, e1, i0, i0s, i1, i1s, L, Wp, bp, N, act, outp)

    // ---------------- user model: item aggregation -------------------------
    GEMM(AM_GATHER2, 4, 128, nb, nb, item_emb, rate_emb, uitem, 2, uitem + 1, 2, 0,
         P(24), P(25), E1, ACT_LEAKY, bufA);                           // g_v L1
    GEMM(AM_DIRECT, 4, 64, bufA, nb, nf, nf, ni, 0, ni, 0, 0,
         P(26), P(27), E1, ACT_NONE, bufB);                            // x_ia
    GEMM(AM_GATHERC, 4, 64, nb, nb, user_emb, nf, uitem, 2, uids, 1, LI_,
         P(42), P(43), E1, ACT_NONE, bufC);                            // w1(p_i)
    GEMM(AM_DIRECT, 4, 64, bufB, nb, nf, nf, ni, 0, ni, 0, 0,
         P(42), P(43), E1, ACT_NONE, bufA);                            // w1(x_ia)
    GEMM(AM_CONCAT2, 4, 128, bufA, bufC, nf, nf, ni, 0, ni, 0, 0,
         P(49), P(50), E1, ACT_LEAKY, bufD);                           // att hidden
    dot_score_kernel<<<(E1 + 255) / 256, 256, 0, stream>>>(bufD, P(51), P(52), E1, 64, score);
    attnorm_wsum_kernel<<<B_, 64, 0, stream>>>(score, uitem, 2, bufB, LI_, g_pre);
    GEMM(AM_DIRECT, 4, 64, g_pre, nb, nf, nf, ni, 0, ni, 0, 0,
         P(7), P(8), B_, ACT_RELU, h_iI);                              // h_iI

    // ---------------- user model: social aggregation -----------------------
    GEMM(AM_GATHER2, 4, 128, nb, nb, item_emb, rate_emb, uuitem, 2, uuitem + 1, 2, 0,
         P(24), P(25), E2, ACT_LEAKY, bufA);                           // g_v L1
    GEMM(AM_DIRECT, 4, 64, bufA, nb, nf, nf, ni, 0, ni, 0, 0,
         P(26), P(27), E2, ACT_NONE, bufB);                            // x_ia_s
    GEMM(AM_DIRECT, 4, 64, bufB, nb, nf, nf, ni, 0, ni, 0, 0,
         P(44), P(45), E2, ACT_NONE, bufA);                            // w4(x_ia_s)
    GEMM(AM_GATHERC, 4, 64, nb, nb, user_emb, nf, uuitem, 2, uuser, 1, LI_,
         P(44), P(45), E2, ACT_NONE, bufC);                            // w4(p_i_s)
    GEMM(AM_CONCAT2, 4, 128, bufA, bufC, nf, nf, ni, 0, ni, 0, 0,
         P(53), P(54), E2, ACT_LEAKY, bufD);
    dot_score_kernel<<<(E2 + 255) / 256, 256, 0, stream>>>(bufD, P(55), P(56), E2, 64, score);
    attnorm_wsum_kernel<<<E3, 64, 0, stream>>>(score, uuitem, 2, bufB, LI_, g_pre);
    GEMM(AM_DIRECT, 4, 64, g_pre, nb, nf, nf, ni, 0, ni, 0, 0,
         P(9), P(10), E3, ACT_RELU, h_oI);                             // h_oI
    GEMM(AM_DIRECT, 4, 64, h_oI, nb, nf, nf, ni, 0, ni, 0, 0,
         P(46), P(47), E3, ACT_NONE, bufA);                            // w5(h_oI)
    GEMM(AM_GATHER1, 4, 64, nb, nb, user_emb, nf, uuser, 1, ni, 0, 0,
         P(46), P(47), E3, ACT_NONE, bufC);                            // w5(ue)
    GEMM(AM_CONCAT2, 4, 128, bufA, bufC, nf, nf, ni, 0, ni, 0, 0,
         P(57), P(58), E3, ACT_LEAKY, bufD);
    dot_score_kernel<<<(E3 + 255) / 256, 256, 0, stream>>>(bufD, P(59), P(60), E3, 64, score);
    attnorm_wsum_kernel<<<B_, 64, 0, stream>>>(score, uuser, 1, h_oI, LU_, g_pre);
    GEMM(AM_DIRECT, 4, 64, g_pre, nb, nf, nf, ni, 0, ni, 0, 0,
         P(11), P(12), B_, ACT_RELU, h_iS);                            // h_iS

    // combine: relu(relu([h_iI|h_iS] @ W1 + b1) @ W2 + b2)
    GEMM(AM_CONCAT2, 8, 128, h_iI, h_iS, nf, nf, ni, 0, ni, 0, 0,
         P(15), P(16), B_, ACT_RELU, h_hid);
    GEMM(AM_DIRECT, 4, 128, h_hid, nb, nf, nf, ni, 0, ni, 0, 0,
         P(17), P(18), B_, ACT_RELU, h_fin);                           // h

    // ---------------- item model -------------------------------------------
    GEMM(AM_GATHER2, 4, 128, nb, nb, user_emb, rate_emb, iuser, 2, iuser + 1, 2, 0,
         P(20), P(21), E3, ACT_LEAKY, bufA);                           // g_u L1
    GEMM(AM_DIRECT, 4, 64, bufA, nb, nf, nf, ni, 0, ni, 0, 0,
         P(22), P(23), E3, ACT_NONE, bufB);                            // f_jt
    GEMM(AM_DIRECT, 4, 64, bufB, nb, nf, nf, ni, 0, ni, 0, 0,
         P(28), P(29), E3, ACT_NONE, bufA);                            // w1f
    GEMM(AM_GATHERC, 4, 64, nb, nb, item_emb, nf, iuser, 2, iids, 1, LU_,
         P(28), P(29), E3, ACT_NONE, bufC);                            // w1(q_j)
    GEMM(AM_CONCAT2, 4, 128, bufA, bufC, nf, nf, ni, 0, ni, 0, 0,
         P(31), P(32), E3, ACT_LEAKY, bufD);
    dot_score_kernel<<<(E3 + 255) / 256, 256, 0, stream>>>(bufD, P(33), P(34), E3, 64, score);
    attnorm_wsum_kernel<<<B_, 64, 0, stream>>>(score, iuser, 2, bufA, LU_, g_pre);
    GEMM(AM_DIRECT, 4, 64, g_pre, nb, nf, nf, ni, 0, ni, 0, 0,
         P(13), P(14), B_, ACT_RELU, z_fin);                           // z

    // ---------------- rating head ------------------------------------------
    GEMM(AM_RATE, 4, 896, h_fin, z_fin, nf, enc_emb, ni, 0, review, 1, 0,
         P(36), P(37), B_, ACT_RELU, r1);                              // 896->64
    GEMM(AM_DIRECT, 1, 64, r1, nb, nf, nf, ni, 0, ni, 0, 0,
         P(38), P(39), B_, ACT_RELU, r2);                              // 64->16
    dot_score_kernel<<<(B_ + 255) / 256, 256, 0, stream>>>(r2, P(40), P(41), B_, 16,
                                                           (float*)d_out);  // 16->1
#undef GEMM
}
